// APPNP_50946902065479
// MI455X (gfx1250) — matrix operand order repaired and checked by reading, not compiled, not run on hardware
//
#include <hip/hip_runtime.h>
#include <hip/hip_bf16.h>

typedef __attribute__((ext_vector_type(16))) __bf16 v16bf;
typedef __attribute__((ext_vector_type(8)))  float  v8f;

#define IN_FEATS 256
#define N_HID    128
#define N_CLS    16
#define ALPHA    0.1f
#define K_ITERS  10

// ---------------------------------------------------------------------------
// Convert the three weight matrices to transposed bf16: Wt[n*K + k] = W[k*N + n]
// so that the WMMA B-fragment K-pairs (k, k+1) are contiguous 32-bit loads.
// ---------------------------------------------------------------------------
__global__ void appnp_prep_weights(const float* __restrict__ W0,
                                   const float* __restrict__ W1,
                                   const float* __restrict__ W2,
                                   __bf16* __restrict__ W0t,
                                   __bf16* __restrict__ W1t,
                                   __bf16* __restrict__ W2t) {
    int t = blockIdx.x * blockDim.x + threadIdx.x;
    if (t < IN_FEATS * N_HID) {               // W0: [256][128] -> W0t: [128][256]
        int n = t / IN_FEATS, k = t % IN_FEATS;
        W0t[t] = (__bf16)W0[(size_t)k * N_HID + n];
    }
    if (t < N_HID * N_HID) {                  // W1: [128][128] -> W1t: [128][128]
        int n = t / N_HID, k = t % N_HID;
        W1t[t] = (__bf16)W1[(size_t)k * N_HID + n];
    }
    if (t < N_HID * N_CLS) {                  // W2: [128][16] -> W2t: [16][128]
        int n = t / N_HID, k = t % N_HID;
        W2t[t] = (__bf16)W2[(size_t)k * N_CLS + n];
    }
}

// ---------------------------------------------------------------------------
// In-degree via float atomics (result reused as norm buffer).
// ---------------------------------------------------------------------------
__global__ void appnp_degree(const int* __restrict__ dst, float* __restrict__ deg, int E) {
    int t = blockIdx.x * blockDim.x + threadIdx.x;
    if (t < E) atomicAdd(&deg[dst[t]], 1.0f);
}

__global__ void appnp_norm(float* __restrict__ nrm, int N) {
    int t = blockIdx.x * blockDim.x + threadIdx.x;
    if (t < N) nrm[t] = rsqrtf(fmaxf(nrm[t], 1.0f));
}

// ---------------------------------------------------------------------------
// Fused 3-layer MLP. One block = 256 threads = 8 wave32s, handles 16 node rows.
// Layer0/1: each wave owns one 16-wide column tile (8 waves x 16 = 128 cols).
// Layer2 (16 output classes): wave 0 only. bf16 WMMA, f32 accumulate.
// Fragment layouts per CDNA5 ISA 7.12.2.
// ---------------------------------------------------------------------------
__global__ __launch_bounds__(256)
void appnp_mlp(const float*  __restrict__ X,
               const __bf16* __restrict__ W0t,
               const float*  __restrict__ b0,
               const __bf16* __restrict__ W1t,
               const float*  __restrict__ b1,
               const __bf16* __restrict__ W2t,
               const float*  __restrict__ b2,
               float* __restrict__ H0,
               int nNodes) {
    __shared__ __bf16 sA [16][IN_FEATS + 4];   // feature tile (padded: stride 260)
    __shared__ __bf16 sH [16][N_HID   + 4];    // layer-0 activations (bf16)
    __shared__ __bf16 sH2[16][N_HID   + 4];    // layer-1 activations (bf16)

    const int tid  = threadIdx.x;
    const int lane = tid & 31;
    const int wave = tid >> 5;
    const int m16  = lane & 15;   // M (or N) index within the 16x16 tile
    const int hi   = lane >> 4;   // half-wave selector
    const int row0 = blockIdx.x * 16;

    // Stage 16x256 feature tile as bf16 in LDS.
    for (int idx = tid; idx < 16 * IN_FEATS; idx += 256) {
        int r = idx >> 8, c = idx & (IN_FEATS - 1);
        float v = (row0 + r < nNodes) ? X[(size_t)(row0 + r) * IN_FEATS + c] : 0.0f;
        sA[r][c] = (__bf16)v;
    }
    __syncthreads();

    const int ng = wave * 16 + m16;            // global output column for layers 0/1
    __builtin_prefetch(&W0t[(size_t)ng * IN_FEATS], 0, 0);

    // ---- layer 0: [16x256] @ [256x128] ----
    v8f acc = {};
    #pragma unroll
    for (int kk = 0; kk < IN_FEATS / 32; ++kk) {
        v16bf a, b;
        #pragma unroll
        for (int i = 0; i < 8; ++i) {
            #pragma unroll
            for (int s = 0; s < 2; ++s) {
                int ka = kk * 32 + (i >> 2) * 16 + hi * 8 + (i & 3) * 2 + s;  // A: 16x32 layout
                a[2 * i + s] = sA[m16][ka];
                int kb = kk * 32 + (i >> 2) * 16 + hi * 8 + (i & 3) * 2 + s;  // B: 32x16 layout
                b[2 * i + s] = W0t[(size_t)ng * IN_FEATS + kb];
            }
        }
        acc = __builtin_amdgcn_wmma_f32_16x16x32_bf16(false, a, false, b,
                                                      (short)0, acc, false, false);
    }
    {
        float bias = b0[ng];
        #pragma unroll
        for (int r = 0; r < 8; ++r) {
            float v = acc[r] + bias;
            sH[hi * 8 + r][ng] = (__bf16)(v > 0.0f ? v : 0.0f);   // ReLU
        }
    }
    __syncthreads();

    // ---- layer 1: [16x128] @ [128x128] ----
    v8f acc1 = {};
    #pragma unroll
    for (int kk = 0; kk < N_HID / 32; ++kk) {
        v16bf a, b;
        #pragma unroll
        for (int i = 0; i < 8; ++i) {
            #pragma unroll
            for (int s = 0; s < 2; ++s) {
                int ka = kk * 32 + (i >> 2) * 16 + hi * 8 + (i & 3) * 2 + s;
                a[2 * i + s] = sH[m16][ka];
                int kb = kk * 32 + (i >> 2) * 16 + hi * 8 + (i & 3) * 2 + s;
                b[2 * i + s] = W1t[(size_t)ng * N_HID + kb];
            }
        }
        acc1 = __builtin_amdgcn_wmma_f32_16x16x32_bf16(false, a, false, b,
                                                       (short)0, acc1, false, false);
    }
    {
        float bias = b1[ng];
        #pragma unroll
        for (int r = 0; r < 8; ++r) {
            float v = acc1[r] + bias;
            sH2[hi * 8 + r][ng] = (__bf16)(v > 0.0f ? v : 0.0f);  // ReLU
        }
    }
    __syncthreads();

    // ---- layer 2: [16x128] @ [128x16] -> global H0 (wave 0 only) ----
    if (wave == 0) {
        v8f acc2 = {};
        #pragma unroll
        for (int kk = 0; kk < N_HID / 32; ++kk) {
            v16bf a, b;
            #pragma unroll
            for (int i = 0; i < 8; ++i) {
                #pragma unroll
                for (int s = 0; s < 2; ++s) {
                    int ka = kk * 32 + (i >> 2) * 16 + hi * 8 + (i & 3) * 2 + s;
                    a[2 * i + s] = sH2[m16][ka];
                    int kb = kk * 32 + (i >> 2) * 16 + hi * 8 + (i & 3) * 2 + s;
                    b[2 * i + s] = W2t[(size_t)m16 * N_HID + kb];
                }
            }
            acc2 = __builtin_amdgcn_wmma_f32_16x16x32_bf16(false, a, false, b,
                                                           (short)0, acc2, false, false);
        }
        float bias = b2[m16];
        #pragma unroll
        for (int r = 0; r < 8; ++r) {
            int row = row0 + hi * 8 + r;
            if (row < nNodes) H0[(size_t)row * N_CLS + m16] = acc2[r] + bias;
        }
    }
}

// ---------------------------------------------------------------------------
// Propagation scatter: 16 consecutive lanes = 16 channels of one edge, so both
// the gather of h[src]*norm[src] and the atomic scatter into agg[dst] coalesce.
// ---------------------------------------------------------------------------
__global__ void appnp_scatter(const float* __restrict__ hin,
                              const float* __restrict__ nrm,
                              const int*   __restrict__ src,
                              const int*   __restrict__ dst,
                              float* __restrict__ agg, int E) {
    unsigned t = blockIdx.x * blockDim.x + threadIdx.x;
    unsigned e = t >> 4;
    int c = t & 15;
    if (e < (unsigned)E) {
        int s = src[e], d = dst[e];
        float v = hin[(size_t)s * N_CLS + c] * nrm[s];
        atomicAdd(&agg[(size_t)d * N_CLS + c], v);
    }
}

// h_next = (1-alpha) * agg * norm[dst] + alpha * h0
__global__ void appnp_blend(const float* __restrict__ agg,
                            const float* __restrict__ nrm,
                            const float* __restrict__ h0,
                            float* __restrict__ out, int N) {
    int t = blockIdx.x * blockDim.x + threadIdx.x;
    if (t < N * N_CLS) {
        int node = t >> 4;
        out[t] = (1.0f - ALPHA) * agg[t] * nrm[node] + ALPHA * h0[t];
    }
}

// ---------------------------------------------------------------------------
extern "C" void kernel_launch(void* const* d_in, const int* in_sizes, int n_in,
                              void* d_out, int out_size, void* d_ws, size_t ws_size,
                              hipStream_t stream) {
    const float* feats = (const float*)d_in[0];
    const int*   src   = (const int*)  d_in[1];
    const int*   dst   = (const int*)  d_in[2];
    const float* W0    = (const float*)d_in[3];
    const float* b0    = (const float*)d_in[4];
    const float* W1    = (const float*)d_in[5];
    const float* b1    = (const float*)d_in[6];
    const float* W2    = (const float*)d_in[7];
    const float* b2    = (const float*)d_in[8];

    const int N = in_sizes[0] / IN_FEATS;   // 100000
    const int E = in_sizes[1];              // 3200000

    // Workspace carve-up (~20 MB total).
    char*  ws  = (char*)d_ws;
    size_t off = 0;
    auto carve = [&](size_t bytes) {
        void* p = ws + off;
        off += (bytes + 255) & ~(size_t)255;
        return p;
    };
    float*  nrm  = (float*)carve((size_t)N * sizeof(float));
    float*  H0   = (float*)carve((size_t)N * N_CLS * sizeof(float));
    float*  Hc   = (float*)carve((size_t)N * N_CLS * sizeof(float));
    float*  Agg  = (float*)carve((size_t)N * N_CLS * sizeof(float));
    __bf16* W0t  = (__bf16*)carve((size_t)IN_FEATS * N_HID * sizeof(__bf16));
    __bf16* W1t  = (__bf16*)carve((size_t)N_HID * N_HID * sizeof(__bf16));
    __bf16* W2t  = (__bf16*)carve((size_t)N_HID * N_CLS * sizeof(__bf16));
    (void)ws_size; (void)n_in; (void)out_size;

    // 1) weights -> transposed bf16
    appnp_prep_weights<<<(IN_FEATS * N_HID + 255) / 256, 256, 0, stream>>>(
        W0, W1, W2, W0t, W1t, W2t);

    // 2) in-degree -> norm = rsqrt(max(deg,1))
    hipMemsetAsync(nrm, 0, (size_t)N * sizeof(float), stream);
    appnp_degree<<<(E + 255) / 256, 256, 0, stream>>>(dst, nrm, E);
    appnp_norm<<<(N + 255) / 256, 256, 0, stream>>>(nrm, N);

    // 3) fused WMMA MLP -> H0
    appnp_mlp<<<(N + 15) / 16, 256, 0, stream>>>(
        feats, W0t, b0, W1t, b1, W2t, b2, H0, N);

    // 4) K propagation steps (ping via Hc; teleport source is always H0)
    const unsigned scatterBlocks = ((unsigned)E * N_CLS + 255) / 256;
    const int blendBlocks = (N * N_CLS + 255) / 256;
    for (int it = 0; it < K_ITERS; ++it) {
        hipMemsetAsync(Agg, 0, (size_t)N * N_CLS * sizeof(float), stream);
        const float* hin = (it == 0) ? H0 : Hc;
        appnp_scatter<<<scatterBlocks, 256, 0, stream>>>(hin, nrm, src, dst, Agg, E);
        float* hout = (it == K_ITERS - 1) ? (float*)d_out : Hc;
        appnp_blend<<<blendBlocks, 256, 0, stream>>>(Agg, nrm, H0, hout, N);
    }
}
